// ExLoss_9096740733605
// MI455X (gfx1250) — matrix operand order repaired and checked by reading, not compiled, not run on hardware
//
#include <hip/hip_runtime.h>
#include <cstdint>

typedef unsigned short u16;
typedef unsigned int u32;
typedef __attribute__((ext_vector_type(16))) __bf16 v16bf;
typedef __attribute__((ext_vector_type(8)))  float  v8f;

static constexpr int B = 2048, C = 16384, D = 512;
static constexpr float MOM = 0.9f;
static constexpr int NCHUNK = 8;
static constexpr int CCHUNK = C / NCHUNK;          // 2048
static constexpr int NWAVES = 8;
static constexpr int BLK = NWAVES * 32;            // 256
static constexpr int MT = 32;                      // rows per block (2 WMMA M-tiles)
static constexpr int COLS_PER_IT = NWAVES * 32;    // 256 (each wave: 2 N-tiles)
static constexpr int NITER = CCHUNK / COLS_PER_IT; // 8
static constexpr int DP = D + 8;                   // LDS row pad (520*2=1040=65*16, keeps 16B align)

// ---------------------------------------------------------------------------
// Split an f32 into bf16 hi + bf16 lo (round-to-nearest-even), so that
// hi + lo ~= f with ~2^-16 relative error. Used for the 3-term WMMA GEMM.
// ---------------------------------------------------------------------------
__global__ void cvt_split_bf16(const float* __restrict__ src,
                               u16* __restrict__ hi, u16* __restrict__ lo, int n) {
  int i = blockIdx.x * blockDim.x + threadIdx.x;
  if (i >= n) return;
  float f = src[i];
  u32 u = __float_as_uint(f);
  u32 rh = u + 0x7FFFu + ((u >> 16) & 1u);
  u16 h = (u16)(rh >> 16);
  float fh = __uint_as_float((u32)h << 16);
  float rem = f - fh;
  u32 u2 = __float_as_uint(rem);
  u32 rl = u2 + 0x7FFFu + ((u2 >> 16) & 1u);
  hi[i] = h;
  lo[i] = (u16)(rl >> 16);
}

// ---------------------------------------------------------------------------
// V_new: exact (deterministic) equivalent of the sequential momentum scan.
// One block per class c: find samples with target==c in ascending batch order
// and apply the recurrence v = m*v + (1-m)*x along D. No atomics on V_new.
// ---------------------------------------------------------------------------
__global__ __launch_bounds__(256) void vnew_update(const float* __restrict__ V,
                                                   const float* __restrict__ X,
                                                   const int* __restrict__ tgt,
                                                   float* __restrict__ Vnew) {
  const int c = blockIdx.x;
  const int tid = threadIdx.x;
  __shared__ int st[B];        // 8 KB
  __shared__ int midx[64];
  __shared__ int mcount;
  if (tid == 0) mcount = 0;
  for (int i = tid; i < B; i += 256) st[i] = tgt[i];
  __syncthreads();
  for (int i = tid; i < B; i += 256)
    if (st[i] == c) { int p = atomicAdd(&mcount, 1); if (p < 64) midx[p] = i; }
  __syncthreads();
  int k = mcount < 64 ? mcount : 64;
  if (tid == 0 && k > 1) { // insertion sort -> ascending batch order
    for (int a = 1; a < k; ++a) {
      int key = midx[a]; int b = a - 1;
      while (b >= 0 && midx[b] > key) { midx[b + 1] = midx[b]; --b; }
      midx[b + 1] = key;
    }
  }
  __syncthreads();
  for (int d = tid; d < D; d += 256) {
    float val = V[(size_t)c * D + d];
    for (int j = 0; j < k; ++j)
      val = MOM * val + (1.0f - MOM) * X[(size_t)midx[j] * D + d];
    Vnew[(size_t)c * D + d] = val;
  }
}

// ---------------------------------------------------------------------------
// WMMA fragment builders (CDNA5 ISA 7.12.2 layouts, wave32).
// A (16-bit, 16x32): lane m -> row m; K chunks {khalf*8+0..7, 16+khalf*8+0..7}
// B (16-bit, 32x16): lane n -> col n; K chunks {khalf*8+0..7, 16+khalf*8+0..7}
// ---------------------------------------------------------------------------
__device__ __forceinline__ v16bf frag_a(const u16* p /*row base*/, int kbase, int khalf) {
  union { uint4 u[2]; v16bf v; } t;
  const u16* q = p + kbase + khalf * 8;
  t.u[0] = *(const uint4*)(q);
  t.u[1] = *(const uint4*)(q + 16);
  return t.v;
}
__device__ __forceinline__ v16bf frag_b(const u16* p /*col(V-row) base*/, int kbase, int khalf) {
  union { uint4 u[2]; v16bf v; } t;
  const u16* q = p + kbase + khalf * 8;
  t.u[0] = *(const uint4*)(q);
  t.u[1] = *(const uint4*)(q + 16);
  return t.v;
}

#define WMMA_BF16(Acc, Af, Bf) \
  Acc = __builtin_amdgcn_wmma_f32_16x16x32_bf16(false, Af, false, Bf, (short)0, Acc, false, false)

// ---------------------------------------------------------------------------
// Fused GEMM (logits = X @ V^T) + per-row sum(exp(logit)) + target-logit grab.
// grid = (B/32, NCHUNK), block = 256 (8 waves).
// Register blocking: 2 M-tiles x 2 N-tiles per wave -> 12 WMMAs per 16 b128
// loads; each B fragment feeds two M-tiles (halves V/L2 traffic).
// 3 bf16 WMMAs per term-pair (hi*hi + hi*lo + lo*hi) ~= f32 accuracy at
// bf16-WMMA throughput. Logits bounded (|l| < ~16) -> no max-shift needed.
// ---------------------------------------------------------------------------
__global__ __launch_bounds__(256) void gemm_ce_partial(
    const u16* __restrict__ Xhi, const u16* __restrict__ Xlo,
    const u16* __restrict__ Vhi, const u16* __restrict__ Vlo,
    const int* __restrict__ tgt,
    float* __restrict__ Spart, float* __restrict__ TgtLog) {
  __shared__ u16 sXh[MT][DP];             // 32.5 KB
  __shared__ u16 sXl[MT][DP];             // 32.5 KB
  __shared__ float sLog[MT][COLS_PER_IT]; // 32 KB
  __shared__ float sRed[MT][8];
  __shared__ float sTgtRed[MT][8];
  __shared__ int sTgt[MT];

  const int tid = threadIdx.x;
  const int wave = tid >> 5, lane = tid & 31;
  const int lrow = lane & 15, khalf = lane >> 4;
  const int rowg = blockIdx.x * MT;
  const int chunkBase = blockIdx.y * CCHUNK;

  // Stage the 32-row X tile (bf16 hi/lo) into LDS.
  for (int u = tid; u < MT * (D / 8); u += BLK) {
    int r = u >> 6, cp = u & 63;
    *(uint4*)&sXh[r][cp * 8] = ((const uint4*)(Xhi + (size_t)(rowg + r) * D))[cp];
    *(uint4*)&sXl[r][cp * 8] = ((const uint4*)(Xlo + (size_t)(rowg + r) * D))[cp];
  }
  if (tid < MT) sTgt[tid] = tgt[rowg + tid];
  __syncthreads();

  const int myrow = tid >> 3, mycol0 = tid & 7;
  float sume = 0.0f, tcap = -3.0e38f;

  for (int it = 0; it < NITER; ++it) {
    const int colW = chunkBase + it * COLS_PER_IT + wave * 32;
    const u16* vh0 = Vhi + (size_t)(colW + lrow) * D;
    const u16* vl0 = Vlo + (size_t)(colW + lrow) * D;
    const u16* vh1 = vh0 + (size_t)16 * D;
    const u16* vl1 = vl0 + (size_t)16 * D;
    if (it + 1 < NITER) { // gfx1250 global_prefetch_b8 for the next V tile
      __builtin_prefetch(vh0 + (size_t)COLS_PER_IT * D, 0, 0);
      __builtin_prefetch(vl0 + (size_t)COLS_PER_IT * D, 0, 0);
      __builtin_prefetch(vh1 + (size_t)COLS_PER_IT * D, 0, 0);
      __builtin_prefetch(vl1 + (size_t)COLS_PER_IT * D, 0, 0);
    }
    v8f a00 = {}, a01 = {}, a10 = {}, a11 = {};
#pragma unroll 2
    for (int kk = 0; kk < D; kk += 32) {
      v16bf ah0 = frag_a(&sXh[lrow][0], kk, khalf);
      v16bf al0 = frag_a(&sXl[lrow][0], kk, khalf);
      v16bf ah1 = frag_a(&sXh[16 + lrow][0], kk, khalf);
      v16bf al1 = frag_a(&sXl[16 + lrow][0], kk, khalf);
      v16bf bh0 = frag_b(vh0, kk, khalf);
      v16bf bl0 = frag_b(vl0, kk, khalf);
      v16bf bh1 = frag_b(vh1, kk, khalf);
      v16bf bl1 = frag_b(vl1, kk, khalf);
      WMMA_BF16(a00, ah0, bh0); WMMA_BF16(a00, ah0, bl0); WMMA_BF16(a00, al0, bh0);
      WMMA_BF16(a01, ah0, bh1); WMMA_BF16(a01, ah0, bl1); WMMA_BF16(a01, al0, bh1);
      WMMA_BF16(a10, ah1, bh0); WMMA_BF16(a10, ah1, bl0); WMMA_BF16(a10, al1, bh0);
      WMMA_BF16(a11, ah1, bh1); WMMA_BF16(a11, ah1, bl1); WMMA_BF16(a11, al1, bh1);
    }
    // C/D layout: VGPR r -> row r + khalf*8, col = lane&15
#pragma unroll
    for (int r = 0; r < 8; ++r) {
      const int mr = r + khalf * 8;
      const int wc = wave * 32 + lrow;
      sLog[mr][wc]           = a00[r];
      sLog[mr][wc + 16]      = a01[r];
      sLog[16 + mr][wc]      = a10[r];
      sLog[16 + mr][wc + 16] = a11[r];
    }
    __syncthreads();

    const int cb = chunkBase + it * COLS_PER_IT;
#pragma unroll
    for (int j = 0; j < COLS_PER_IT / 8; ++j) {
      int cc = mycol0 + j * 8;
      float l = sLog[myrow][cc];
      sume += __expf(l);
      if (cb + cc == sTgt[myrow]) tcap = l;
    }
    __syncthreads();
  }

  sRed[myrow][mycol0] = sume;
  sTgtRed[myrow][mycol0] = tcap;
  __syncthreads();
  if (tid < MT) {
    float S = 0.0f, tg = -3.0e38f;
#pragma unroll
    for (int j = 0; j < 8; ++j) { S += sRed[tid][j]; tg = fmaxf(tg, sTgtRed[tid][j]); }
    const int rg = rowg + tid;
    Spart[rg * NCHUNK + blockIdx.y] = S;
    const int t = sTgt[tid];
    if (t >= chunkBase && t < chunkBase + CCHUNK) TgtLog[rg] = tg;
  }
}

// ---------------------------------------------------------------------------
// loss = mean_i( log(sum_c exp(l_ic)) - l_{i,t_i} )
// ---------------------------------------------------------------------------
__global__ __launch_bounds__(256) void ce_finalize(const float* __restrict__ Spart,
                                                   const float* __restrict__ TgtLog,
                                                   float* __restrict__ out) {
  __shared__ float red[256];
  const int tid = threadIdx.x;
  float local = 0.0f;
  for (int i = tid; i < B; i += 256) {
    float S = 0.0f;
#pragma unroll
    for (int ch = 0; ch < NCHUNK; ++ch) S += Spart[i * NCHUNK + ch];
    local += logf(S) - TgtLog[i];
  }
  red[tid] = local;
  __syncthreads();
  for (int s = 128; s > 0; s >>= 1) {
    if (tid < s) red[tid] += red[tid + s];
    __syncthreads();
  }
  if (tid == 0) out[0] = red[0] / (float)B;
}

// ---------------------------------------------------------------------------
extern "C" void kernel_launch(void* const* d_in, const int* in_sizes, int n_in,
                              void* d_out, int out_size, void* d_ws, size_t ws_size,
                              hipStream_t stream) {
  (void)in_sizes; (void)n_in; (void)out_size; (void)ws_size;
  const float* x   = (const float*)d_in[0];  // [B, D]
  const int*   tgt = (const int*)d_in[1];    // [B]
  const float* V   = (const float*)d_in[2];  // [C, D]
  float* out  = (float*)d_out;               // [0] = loss
  float* Vnew = out + 1;                     // [C, D]

  char* ws = (char*)d_ws;
  size_t off = 0;
  auto take = [&](size_t bytes) -> char* {
    char* p = ws + off;
    off = (off + bytes + 255) & ~(size_t)255;
    return p;
  };
  u16* Xhi = (u16*)take((size_t)B * D * 2);   // 2 MB
  u16* Xlo = (u16*)take((size_t)B * D * 2);   // 2 MB
  u16* Vhi = (u16*)take((size_t)C * D * 2);   // 16 MB
  u16* Vlo = (u16*)take((size_t)C * D * 2);   // 16 MB
  float* Spart  = (float*)take((size_t)B * NCHUNK * 4);
  float* TgtLog = (float*)take((size_t)B * 4);

  cvt_split_bf16<<<(B * D + 255) / 256, 256, 0, stream>>>(x, Xhi, Xlo, B * D);
  cvt_split_bf16<<<(C * D + 255) / 256, 256, 0, stream>>>(V, Vhi, Vlo, C * D);
  vnew_update<<<C, 256, 0, stream>>>(V, x, tgt, Vnew);
  gemm_ce_partial<<<dim3(B / MT, NCHUNK), BLK, 0, stream>>>(Xhi, Xlo, Vhi, Vlo, tgt, Spart, TgtLog);
  ce_finalize<<<1, 256, 0, stream>>>(Spart, TgtLog, out);
}
